// Decoder_56255481643086
// MI455X (gfx1250) — compile-verified, matrix operand order checked by reading, not run on hardware
//
#include <hip/hip_runtime.h>

typedef __attribute__((ext_vector_type(16))) _Float16 v16h;
typedef __attribute__((ext_vector_type(8)))  float    v8f;

#define N_AGENTS 10
#define N_EN 11
#define N_AL 9
#define OWNF 10
#define NF_EN 8
#define NF_AL 8
#define NNA 6
#define ID_LEN 8
#define EMB 64
#define HEADS 4
#define DEPTH 2
#define BSZ 4096
#define BTOT (BSZ * N_AGENTS)          // 40960
#define OBS_DIM 170
#define IN_DIM 195
#define T_PAD 32
#define T_REAL 23

// workspace offsets (in halves), all 32B aligned
#define WS_WQ   0        // 2 x 64x256
#define WS_WK   32768
#define WS_WV   65536
#define WS_WU   98304    // 2 x 256x64
#define WS_WF1  131072   // 2 x 64x256
#define WS_WF2  163840   // 2 x 256x64
#define WS_WP   196608   // 64x64
#define WS_WA1  200704   // 128x128
#define WS_WB1  217088   // 128x128

// ---------- WMMA helpers ----------
__device__ __forceinline__ v8f wmma_f16(v16h a, v16h b, v8f c) {
  return __builtin_amdgcn_wmma_f32_16x16x32_f16(false, a, false, b, (short)0, c,
                                                false, false);
}

// A-fragment (16x32 f16 tile) from a row-major LDS f16 buffer.
// lane L: row = mt*16 + (L&15); k = kt*32 + { g..g+7 , 16+g..16+g+7 }, g=(L>>4)*8
__device__ __forceinline__ v16h afrag_h(const _Float16* buf, int stride, int mt,
                                        int kt, int lane) {
  const _Float16* p = buf + (mt * 16 + (lane & 15)) * stride + kt * 32 + ((lane >> 4) << 3);
  v16h r;
#pragma unroll
  for (int i = 0; i < 8; ++i) { r[i] = p[i]; r[8 + i] = p[16 + i]; }
  return r;
}

// A-fragment converted on the fly from a row-major f32 LDS buffer (softmax probs)
__device__ __forceinline__ v16h afrag_f(const float* buf, int stride, int mt,
                                        int kt, int lane) {
  const float* p = buf + (mt * 16 + (lane & 15)) * stride + kt * 32 + ((lane >> 4) << 3);
  v16h r;
#pragma unroll
  for (int i = 0; i < 8; ++i) { r[i] = (_Float16)p[i]; r[8 + i] = (_Float16)p[16 + i]; }
  return r;
}

// B-fragment from pre-swizzled global weights: one contiguous 32B load per lane
__device__ __forceinline__ v16h bfrag_g(const _Float16* w, int KT, int kt, int nt,
                                        int lane) {
  return *(const v16h*)(w + ((((nt * KT) + kt) * 32 + lane) << 4));
}

// B-fragment (K x N tile) from a row-major LDS f16 buffer (used for V)
__device__ __forceinline__ v16h bfrag_row(const _Float16* buf, int stride, int kt,
                                          int nt, int lane) {
  int n = nt * 16 + (lane & 15);
  const _Float16* p = buf + (kt * 32 + ((lane >> 4) << 3)) * stride + n;
  v16h r;
#pragma unroll
  for (int i = 0; i < 8; ++i) { r[i] = p[i * stride]; r[8 + i] = p[(16 + i) * stride]; }
  return r;
}

// D (v8f) stores: lane L holds col n=(L&15), rows m0..m0+7, m0 = mt*16+(L>>4)*8
__device__ __forceinline__ void store_h(_Float16* buf, int stride, int mt, int nt,
                                        int lane, v8f d) {
  int n = nt * 16 + (lane & 15);
  int m0 = mt * 16 + ((lane >> 4) << 3);
#pragma unroll
  for (int r = 0; r < 8; ++r) buf[(m0 + r) * stride + n] = (_Float16)d[r];
}

__device__ __forceinline__ void store_f_scaled(float* buf, int stride, int mt, int nt,
                                               int lane, v8f d, float s) {
  int n = nt * 16 + (lane & 15);
  int m0 = mt * 16 + ((lane >> 4) << 3);
#pragma unroll
  for (int r = 0; r < 8; ++r) buf[(m0 + r) * stride + n] = d[r] * s;
}

__device__ __forceinline__ void store_f_bias(float* buf, int stride, int mt, int nt,
                                             int lane, v8f d, const float* bias) {
  int n = nt * 16 + (lane & 15);
  int m0 = mt * 16 + ((lane >> 4) << 3);
  float b = bias[n];
#pragma unroll
  for (int r = 0; r < 8; ++r) buf[(m0 + r) * stride + n] = d[r] + b;
}

__device__ __forceinline__ void store_h_relu_bias(_Float16* buf, int stride, int mt,
                                                  int nt, int lane, v8f d,
                                                  const float* bias) {
  int n = nt * 16 + (lane & 15);
  int m0 = mt * 16 + ((lane >> 4) << 3);
  float b = bias[n];
#pragma unroll
  for (int r = 0; r < 8; ++r) {
    float v = d[r] + b;
    buf[(m0 + r) * stride + n] = (_Float16)(v > 0.f ? v : 0.f);
  }
}

// ---------- weight swizzle: f32 row-major (K x N) -> f16 WMMA-B layout ----------
__global__ void swizzle_weights(const float* __restrict__ src, _Float16* __restrict__ dst,
                                int K, int N, int count) {
  int idx = blockIdx.x * blockDim.x + threadIdx.x;
  int per = K * N;
  if (idx >= per * count) return;
  int m = idx / per, r = idx - m * per;
  int KT = K >> 5;
  int h = r & 15;
  int lane = (r >> 4) & 31;
  int rest = r >> 9;
  int tk = rest % KT;
  int tn = rest / KT;
  int k = tk * 32 + ((h < 8 ? h : h + 8) + ((lane >> 4) << 3));
  int n = tn * 16 + (lane & 15);
  dst[idx] = (_Float16)src[(size_t)m * per + (size_t)k * N + n];
}

// ---------- LayerNorm: x = LN(x + y); re-zero pad rows; refresh f16 mirror ----------
__device__ __forceinline__ void ln_rows(float* xf, const float* yf, _Float16* xh,
                                        const float* sc, const float* bi, int tid) {
  if (tid < T_PAD) {
    float* xr = xf + tid * EMB;
    const float* yr = yf + tid * EMB;
    if (tid < T_REAL) {
      float m = 0.f;
      for (int i = 0; i < EMB; ++i) m += xr[i] + yr[i];
      m *= (1.f / EMB);
      float v = 0.f;
      for (int i = 0; i < EMB; ++i) { float dd = xr[i] + yr[i] - m; v += dd * dd; }
      v *= (1.f / EMB);
      float inv = rsqrtf(v + 1e-5f);
      for (int i = 0; i < EMB; ++i) {
        float o = (xr[i] + yr[i] - m) * inv * sc[i] + bi[i];
        xr[i] = o;
        xh[tid * EMB + i] = (_Float16)o;
      }
    } else {
      for (int i = 0; i < EMB; ++i) { xr[i] = 0.f; xh[tid * EMB + i] = (_Float16)0.f; }
    }
  }
}

// ---------- fused forward: 1 workgroup (4 waves) per sample ----------
__global__ __launch_bounds__(128) void mac_forward(
    const float* __restrict__ inputs, const float* __restrict__ hidden_state,
    const float* __restrict__ skill_hidden, const float* __restrict__ W_own,
    const float* __restrict__ b_own, const float* __restrict__ W_en,
    const float* __restrict__ b_en, const float* __restrict__ W_al,
    const float* __restrict__ b_al, const float* __restrict__ bu,
    const float* __restrict__ ln1_s, const float* __restrict__ ln1_b,
    const float* __restrict__ ln2_s, const float* __restrict__ ln2_b,
    const float* __restrict__ bf1, const float* __restrict__ bf2,
    const float* __restrict__ bp, const float* __restrict__ bb1,
    const float* __restrict__ Wb2, const float* __restrict__ bb2,
    const float* __restrict__ ba1, const float* __restrict__ Wa2,
    const float* __restrict__ ba2, const _Float16* __restrict__ wsh,
    float* __restrict__ out) {
  __shared__ float xf[T_PAD * EMB];      // activation master (f32)
  __shared__ float yf[T_PAD * EMB];      // residual branch / final out
  __shared__ _Float16 xh[T_PAD * EMB];   // f16 mirror of xf
  __shared__ _Float16 qh[T_PAD * EMB];   // per-head Q (reused: attack hidden)
  __shared__ _Float16 kh[T_PAD * EMB];   // per-head K (reused: base hidden)
  __shared__ _Float16 vh[T_PAD * EMB];   // per-head V
  __shared__ _Float16 oh[T_PAD * EMB];   // per-head O / FF chunk / head A-matrix
  __shared__ float sS[T_PAD * T_PAD];    // attention scores / probs

  const int s = blockIdx.x;
  const int tid = threadIdx.x;
  const int lane = tid & 31;
  const int w = tid >> 5;  // wave id 0..3
  const float* inp = inputs + (size_t)s * IN_DIM;

  // ---- token embeddings ----
  float asum = 0.f;
  for (int e = 0; e < N_EN; ++e) asum += inp[OBS_DIM + NNA + e];
  const int idnum = (s % N_AGENTS) + 1;

  for (int i = tid; i < T_PAD * EMB; i += 128) {
    int row = i >> 6, col = i & 63;
    float v = 0.f;
    if (row == 0) {
      v = skill_hidden[col];
    } else if (row == 1) {
      v = b_own[col];
      for (int f = 0; f < OWNF; ++f) v += inp[f] * W_own[f * EMB + col];
      for (int j = 0; j < ID_LEN; ++j)
        v += (float)((idnum >> (7 - j)) & 1) * W_own[(OWNF + j) * EMB + col];
      for (int f = 0; f < NNA; ++f)
        v += inp[OBS_DIM + f] * W_own[(OWNF + ID_LEN + f) * EMB + col];
      v += asum * W_own[(OWNF + ID_LEN + NNA) * EMB + col];
    } else if (row < 2 + N_EN) {
      int e = row - 2;
      v = b_en[col];
      const float* ef = inp + OWNF + e * NF_EN;
      for (int f = 0; f < NF_EN; ++f) v += ef[f] * W_en[f * EMB + col];
      v += inp[OBS_DIM + NNA + e] * W_en[NF_EN * EMB + col];
    } else if (row < 2 + N_EN + N_AL) {
      int a = row - 2 - N_EN;
      v = b_al[col];
      const float* af = inp + OWNF + N_EN * NF_EN + a * NF_AL;
      for (int f = 0; f < NF_AL; ++f) v += af[f] * W_al[f * EMB + col];
    } else if (row == 22) {
      v = hidden_state[(size_t)s * EMB + col];
    }  // rows 23..31 stay zero
    xf[i] = v;
    xh[i] = (_Float16)v;
  }
  __syncthreads();

  // ---- transformer layers ----
  for (int d = 0; d < DEPTH; ++d) {
    const _Float16* Wq = wsh + WS_WQ + d * EMB * 256;
    const _Float16* Wk = wsh + WS_WK + d * EMB * 256;
    const _Float16* Wv = wsh + WS_WV + d * EMB * 256;
    const _Float16* Wu = wsh + WS_WU + d * 256 * EMB;
    const _Float16* Wf1 = wsh + WS_WF1 + d * EMB * 256;
    const _Float16* Wf2 = wsh + WS_WF2 + d * 256 * EMB;

    v8f accY0 = {}, accY1 = {};  // wave w owns output N-tile w, M-tiles 0/1
    for (int h = 0; h < HEADS; ++h) {
      // QKV for head h: wave w computes head-local N-tile w (global nt = h*4+w)
#pragma unroll
      for (int mt = 0; mt < 2; ++mt) {
        v8f cq = {}, ck = {}, cv = {};
#pragma unroll
        for (int kt = 0; kt < 2; ++kt) {
          v16h a = afrag_h(xh, EMB, mt, kt, lane);
          cq = wmma_f16(a, bfrag_g(Wq, 2, kt, h * 4 + w, lane), cq);
          ck = wmma_f16(a, bfrag_g(Wk, 2, kt, h * 4 + w, lane), ck);
          cv = wmma_f16(a, bfrag_g(Wv, 2, kt, h * 4 + w, lane), cv);
        }
        store_h(qh, EMB, mt, w, lane, cq);
        store_h(kh, EMB, mt, w, lane, ck);
        store_h(vh, EMB, mt, w, lane, cv);
      }
      __syncthreads();
      // S = (Q K^T) * 1/8   (EMB^-0.25 applied to q and k => 1/8 on scores)
      {
        int mt = w >> 1, nt = w & 1;
        v8f cs = {};
#pragma unroll
        for (int kt = 0; kt < 2; ++kt) {
          v16h a = afrag_h(qh, EMB, mt, kt, lane);
          v16h bT = afrag_h(kh, EMB, nt, kt, lane);  // K^T B-frag == A-frag pattern
          cs = wmma_f16(a, bT, cs);
        }
        store_f_scaled(sS, T_PAD, mt, nt, lane, cs, 0.125f);
      }
      __syncthreads();
      // row softmax, masking padded key columns
      if (tid < T_PAD) {
        float* r = sS + tid * T_PAD;
        float mx = -1e30f;
        for (int c = 0; c < T_REAL; ++c) mx = fmaxf(mx, r[c]);
        float sum = 0.f;
        for (int c = 0; c < T_REAL; ++c) { float e = __expf(r[c] - mx); r[c] = e; sum += e; }
        float inv = 1.f / sum;
        for (int c = 0; c < T_REAL; ++c) r[c] *= inv;
        for (int c = T_REAL; c < T_PAD; ++c) r[c] = 0.f;
      }
      __syncthreads();
      // O = S @ V ; wave w computes head-local N-tile w
#pragma unroll
      for (int mt = 0; mt < 2; ++mt) {
        v16h a = afrag_f(sS, T_PAD, mt, 0, lane);
        v16h b = bfrag_row(vh, EMB, 0, w, lane);
        v8f c = {};
        c = wmma_f16(a, b, c);
        store_h(oh, EMB, mt, w, lane, c);
      }
      __syncthreads();
      // accY += O_h @ Wu_h (accumulated across heads in registers)
#pragma unroll
      for (int kk = 0; kk < 2; ++kk) {
        v16h b = bfrag_g(Wu, 8, h * 2 + kk, w, lane);
        accY0 = wmma_f16(afrag_h(oh, EMB, 0, kk, lane), b, accY0);
        accY1 = wmma_f16(afrag_h(oh, EMB, 1, kk, lane), b, accY1);
      }
      __syncthreads();
    }
    store_f_bias(yf, EMB, 0, w, lane, accY0, bu + d * EMB);
    store_f_bias(yf, EMB, 1, w, lane, accY1, bu + d * EMB);
    __syncthreads();
    ln_rows(xf, yf, xh, ln1_s + d * EMB, ln1_b + d * EMB, tid);
    __syncthreads();

    // ---- feed-forward, 32-wide hidden chunks, accumulate in registers ----
    v8f accF0 = {}, accF1 = {};
    for (int cc = 0; cc < 8; ++cc) {
      {
        int mt = w >> 1, ntl = w & 1;
        v8f ch = {};
#pragma unroll
        for (int kt = 0; kt < 2; ++kt) {
          v16h a = afrag_h(xh, EMB, mt, kt, lane);
          ch = wmma_f16(a, bfrag_g(Wf1, 2, kt, cc * 2 + ntl, lane), ch);
        }
        store_h_relu_bias(oh, 32, mt, ntl, lane, ch, bf1 + d * 256 + cc * 32);
      }
      __syncthreads();
      {
        v16h b = bfrag_g(Wf2, 8, cc, w, lane);
        accF0 = wmma_f16(afrag_h(oh, 32, 0, 0, lane), b, accF0);
        accF1 = wmma_f16(afrag_h(oh, 32, 1, 0, lane), b, accF1);
      }
      __syncthreads();
    }
    store_f_bias(yf, EMB, 0, w, lane, accF0, bf2 + d * EMB);
    store_f_bias(yf, EMB, 1, w, lane, accF1, bf2 + d * EMB);
    __syncthreads();
    ln_rows(xf, yf, xh, ln2_s + d * EMB, ln2_b + d * EMB, tid);
    __syncthreads();
  }

  // ---- out = x @ Wp + bp -> yf ----
  {
    const _Float16* Wp = wsh + WS_WP;
#pragma unroll
    for (int mt = 0; mt < 2; ++mt) {
      v8f c = {};
#pragma unroll
      for (int kt = 0; kt < 2; ++kt)
        c = wmma_f16(afrag_h(xh, EMB, mt, kt, lane), bfrag_g(Wp, 2, kt, w, lane), c);
      store_f_bias(yf, EMB, mt, w, lane, c, bp);
    }
  }
  __syncthreads();

  float* q_out = out;                             // B x 17
  float* skill_out = out + (size_t)BTOT * 17;     // B x 64
  float* h_out = skill_out + (size_t)BTOT * EMB;  // B x 64
  if (tid < EMB) {
    skill_out[(size_t)s * EMB + tid] = yf[tid];         // out[:,0,:]
    h_out[(size_t)s * EMB + tid] = yf[22 * EMB + tid];  // out[:,-1,:]
  }
  // Shared head A-matrix (16 x 128) in oh:
  //   rows 0..10 = [out[2+e], skill]   (attack head)
  //   row  11    = [out[1],   skill]   (base head)
  //   rows 12..15 zero
  for (int i = tid; i < 16 * 128; i += 128) {
    int e = i >> 7, c = i & 127;
    float v = 0.f;
    if (e < N_EN) v = (c < EMB) ? yf[(2 + e) * EMB + c] : yf[c - EMB];
    else if (e == N_EN) v = (c < EMB) ? yf[EMB + c] : yf[c - EMB];
    oh[i] = (_Float16)v;
  }
  __syncthreads();
  // Two hidden layers from the same A: attack (Wa1) -> qh, base (Wb1) -> kh
  {
    const _Float16* Wa1s = wsh + WS_WA1;
    const _Float16* Wb1s = wsh + WS_WB1;
#pragma unroll
    for (int j2 = 0; j2 < 2; ++j2) {
      int nt = w * 2 + j2;
      v8f ca = {}, cb = {};
#pragma unroll
      for (int kt = 0; kt < 4; ++kt) {
        v16h a = afrag_h(oh, 128, 0, kt, lane);
        ca = wmma_f16(a, bfrag_g(Wa1s, 4, kt, nt, lane), ca);
        cb = wmma_f16(a, bfrag_g(Wb1s, 4, kt, nt, lane), cb);
      }
      store_h_relu_bias(qh, 128, 0, nt, lane, ca, ba1);
      store_h_relu_bias(kh, 128, 0, nt, lane, cb, bb1);
    }
  }
  __syncthreads();
  if (tid < N_EN) {
    float qv = ba2[0];
    for (int j = 0; j < 128; ++j) qv += (float)qh[tid * 128 + j] * Wa2[j];
    q_out[(size_t)s * 17 + 6 + tid] = qv;
  } else if (tid >= 32 && tid < 38) {
    int n = tid - 32;
    float qv = bb2[n];
    for (int j = 0; j < 128; ++j) qv += (float)kh[N_EN * 128 + j] * Wb2[j * 6 + n];
    q_out[(size_t)s * 17 + n] = qv;
  }
}

extern "C" void kernel_launch(void* const* d_in, const int* in_sizes, int n_in,
                              void* d_out, int out_size, void* d_ws, size_t ws_size,
                              hipStream_t stream) {
  (void)in_sizes; (void)n_in; (void)out_size; (void)ws_size;
  _Float16* wsh = (_Float16*)d_ws;

  auto swz = [&](int idx, int off, int K, int N, int cnt) {
    int total = K * N * cnt;
    swizzle_weights<<<(total + 255) / 256, 256, 0, stream>>>(
        (const float*)d_in[idx], wsh + off, K, N, cnt);
  };
  swz(9,  WS_WQ,  64, 256, 2);   // Wq
  swz(10, WS_WK,  64, 256, 2);   // Wk
  swz(11, WS_WV,  64, 256, 2);   // Wv
  swz(12, WS_WU,  256, 64, 2);   // Wu
  swz(18, WS_WF1, 64, 256, 2);   // Wf1
  swz(20, WS_WF2, 256, 64, 2);   // Wf2
  swz(22, WS_WP,  64, 64, 1);    // Wp
  swz(28, WS_WA1, 128, 128, 1);  // Wa1
  swz(24, WS_WB1, 128, 128, 1);  // Wb1

  mac_forward<<<BTOT, 128, 0, stream>>>(
      (const float*)d_in[0],   // inputs
      (const float*)d_in[1],   // hidden_state
      (const float*)d_in[2],   // skill_hidden
      (const float*)d_in[3], (const float*)d_in[4],    // W_own, b_own
      (const float*)d_in[5], (const float*)d_in[6],    // W_en, b_en
      (const float*)d_in[7], (const float*)d_in[8],    // W_al, b_al
      (const float*)d_in[13],                          // bu
      (const float*)d_in[14], (const float*)d_in[15],  // ln1_s, ln1_b
      (const float*)d_in[16], (const float*)d_in[17],  // ln2_s, ln2_b
      (const float*)d_in[19], (const float*)d_in[21],  // bf1, bf2
      (const float*)d_in[23],                          // bp
      (const float*)d_in[25],                          // bb1
      (const float*)d_in[26], (const float*)d_in[27],  // Wb2, bb2
      (const float*)d_in[29],                          // ba1
      (const float*)d_in[30], (const float*)d_in[31],  // Wa2, ba2
      wsh, (float*)d_out);
}
